// MKMMDLoss_73426760892841
// MI455X (gfx1250) — matrix operand order repaired and checked by reading, not compiled, not run on hardware
//
#include <hip/hip_runtime.h>

typedef __attribute__((ext_vector_type(16))) _Float16 v16h;
typedef __attribute__((ext_vector_type(8)))  _Float16 v8h;
typedef __attribute__((ext_vector_type(8)))  float    v8f;
typedef __attribute__((ext_vector_type(4)))  int      v4i;

#define N_ROWS 8192
#define DIM    256
#define BM     128                 // block tile rows
#define BN     256                 // block tile cols
#define BK     32
#define KSTEPS (DIM / BK)          // 8
#define TM     (N_ROWS / BM)       // 64
#define TN     (N_ROWS / BN)       // 32
#define NPARTIAL (3 * TM * TN)     // 6144
#define LDSTRIDE (BK + 8)          // 40 halfs = 80B row stride (16B aligned, bank-spread)

// CDNA5 async global->LDS path (signature confirmed by round-2 stderr:
// (int4 __device__*, int4 __shared__*, i32 offset, i32 cpol)).
#if defined(__has_builtin)
#  if __has_builtin(__builtin_amdgcn_global_load_async_to_lds_b128) && \
      __has_builtin(__builtin_amdgcn_s_wait_asynccnt)
#    define USE_ASYNC_LDS 1
#  endif
#endif
#ifndef USE_ASYNC_LDS
#  define USE_ASYNC_LDS 0
#endif

#if USE_ASYNC_LDS
typedef __attribute__((address_space(1))) v4i gv4i_t;   // global
typedef __attribute__((address_space(3))) v4i sv4i_t;   // LDS
#define COPY_B128(gsrc, ldst)                                                 \
    __builtin_amdgcn_global_load_async_to_lds_b128(                           \
        (gv4i_t*)(void*)const_cast<_Float16*>(gsrc),                          \
        (sv4i_t*)(void*)(ldst), 0, 0)
#define WAIT_ASYNC() __builtin_amdgcn_s_wait_asynccnt(0)
#else
#define COPY_B128(gsrc, ldst) (*(v8h*)(ldst) = *(const v8h*)(gsrc))
#define WAIT_ASYNC() ((void)0)
#endif

// -------------------------------------------------------------------------
// Kernel 1: fp32 -> f16 conversion + per-row squared norms.
// -------------------------------------------------------------------------
__global__ __launch_bounds__(256)
void mmd_prep_kernel(const float* __restrict__ src, const float* __restrict__ tgt,
                     _Float16* __restrict__ Sh, _Float16* __restrict__ Th,
                     float* __restrict__ ns, float* __restrict__ nt) {
    const float* in  = blockIdx.y ? tgt : src;
    _Float16*    out = blockIdx.y ? Th  : Sh;
    float*       nrm = blockIdx.y ? nt  : ns;

    const int wave = threadIdx.x >> 5;
    const int lane = threadIdx.x & 31;
    const int row  = blockIdx.x * 8 + wave;

    const float4* p = (const float4*)(in + (size_t)row * DIM + lane * 8);
    float4 a = p[0];
    float4 b = p[1];

    v8h hv;
    hv[0] = (_Float16)a.x; hv[1] = (_Float16)a.y;
    hv[2] = (_Float16)a.z; hv[3] = (_Float16)a.w;
    hv[4] = (_Float16)b.x; hv[5] = (_Float16)b.y;
    hv[6] = (_Float16)b.z; hv[7] = (_Float16)b.w;
    *(v8h*)(out + (size_t)row * DIM + lane * 8) = hv;

    float sq = a.x*a.x + a.y*a.y + a.z*a.z + a.w*a.w
             + b.x*b.x + b.y*b.y + b.z*b.z + b.w*b.w;
    #pragma unroll
    for (int m = 16; m > 0; m >>= 1) sq += __shfl_xor(sq, m, 32);
    if (lane == 0) nrm[row] = sq;
}

// -------------------------------------------------------------------------
// Kernel 2: fused Gram-tile + exp-kernel reduction.
// 256 threads = 8 waves (2x4); each wave owns a 64x64 C-footprint.
// Double-buffered async global->LDS staging: DMA for K-step ks+1 overlaps
// all 16 WMMAs of K-step ks; s_wait_asynccnt only at the barrier.
// -------------------------------------------------------------------------
__global__ __launch_bounds__(256)
void mmd_main_kernel(const _Float16* __restrict__ Sh, const _Float16* __restrict__ Th,
                     const float* __restrict__ ns, const float* __restrict__ nt,
                     float* __restrict__ partials) {
    const int br = blockIdx.x;       // M tile index (0..63)
    const int bc = blockIdx.y;       // N tile index (0..31)
    const int p  = blockIdx.z;       // pair index
    const int pidx = (p * TN + bc) * TM + br;

    const _Float16 *A, *B;
    const float *nA, *nB;
    if (p == 0)      { A = Sh; B = Sh; nA = ns; nB = ns; }
    else if (p == 1) { A = Th; B = Th; nA = nt; nB = nt; }
    else             { A = Sh; B = Th; nA = ns; nB = nt; }

    float wblk;
    bool  mixed = false;
    if (p < 2) {
        if (br >= 2 * bc + 2) {                 // block entirely below diagonal
            if (threadIdx.x == 0) partials[pidx] = 0.0f;
            return;
        }
        if (br < 2 * bc) { wblk = 2.0f; }       // entirely above diagonal
        else             { wblk = 1.0f; mixed = true; }   // straddles diagonal
    } else {
        wblk = -2.0f;
    }

    __shared__ __align__(16) _Float16 As[2][BM][LDSTRIDE];
    __shared__ __align__(16) _Float16 Bs[2][BN][LDSTRIDE];
    __shared__ float nAs[BM];
    __shared__ float nBs[BN];
    __shared__ float red[8];

    const int tid = threadIdx.x;
    if (tid < BM) nAs[tid] = nA[br * BM + tid];
    nBs[tid] = nB[bc * BN + tid];

    const int wid   = tid >> 5;
    const int lane  = tid & 31;
    const int waveM = wid >> 2;      // 0..1  -> 64 rows each
    const int waveN = wid & 3;       // 0..3  -> 64 cols each
    const int hi    = lane >> 4;     // lane half (K-split per ISA layout)
    const int lr    = lane & 15;

    v8f c[4][4] = {};                // 16 accumulators (4 M-tiles x 4 N-tiles)

    // staging: A -> 2 threads/row (128 rows); B -> 2 rows/thread (256 rows)
    const int gRow = tid >> 1;           // 0..127
    const int gOff = (tid & 1) * 16;     // 0 or 16

    const _Float16* gaBase  = A + (size_t)(br * BM + gRow) * DIM + gOff;
    const _Float16* gbBase0 = B + (size_t)(bc * BN + gRow) * DIM + gOff;
    const _Float16* gbBase1 = B + (size_t)(bc * BN + 128 + gRow) * DIM + gOff;

#define STAGE(ks, buf)                                                        \
    do {                                                                      \
        const int k0_ = (ks) * BK;                                            \
        COPY_B128(gaBase  + k0_,     &As[buf][gRow][gOff]);                   \
        COPY_B128(gaBase  + k0_ + 8, &As[buf][gRow][gOff + 8]);               \
        COPY_B128(gbBase0 + k0_,     &Bs[buf][gRow][gOff]);                   \
        COPY_B128(gbBase0 + k0_ + 8, &Bs[buf][gRow][gOff + 8]);               \
        COPY_B128(gbBase1 + k0_,     &Bs[buf][128 + gRow][gOff]);             \
        COPY_B128(gbBase1 + k0_ + 8, &Bs[buf][128 + gRow][gOff + 8]);         \
    } while (0)

    // prologue: fill buffer 0
    STAGE(0, 0);
    WAIT_ASYNC();
    __syncthreads();

    for (int ks = 0; ks < KSTEPS; ++ks) {
        const int cur = ks & 1;
        if (ks + 1 < KSTEPS) {
            STAGE(ks + 1, cur ^ 1);          // async DMA overlaps compute below
            __builtin_prefetch(gaBase  + (ks + 2) * BK, 0, 3);
            __builtin_prefetch(gbBase0 + (ks + 2) * BK, 0, 3);
            __builtin_prefetch(gbBase1 + (ks + 2) * BK, 0, 3);
        }

        // B fragments: lane n holds K rows (16*hi..+15) of column lr
        v16h bfrag[4];
        #pragma unroll
        for (int tn = 0; tn < 4; ++tn) {
            const int col = waveN * 64 + tn * 16 + lr;
            union { v16h v; v8h h[2]; } u;
            u.h[0] = *(const v8h*)&Bs[cur][col][hi * 16];
            u.h[1] = *(const v8h*)&Bs[cur][col][hi * 16 + 8];
            bfrag[tn] = u.v;
        }
        #pragma unroll
        for (int tm = 0; tm < 4; ++tm) {
            const int row = waveM * 64 + tm * 16 + lr;
            union { v16h v; v8h h[2]; } u;
            u.h[0] = *(const v8h*)&As[cur][row][hi * 16];       // K {0-7} / {16-23}
            u.h[1] = *(const v8h*)&As[cur][row][hi * 16 + 8];   // K {8-15} / {24-31}
            const v16h afrag = u.v;
            #pragma unroll
            for (int tn = 0; tn < 4; ++tn) {
                c[tm][tn] = __builtin_amdgcn_wmma_f32_16x16x32_f16(
                    false, afrag, false, bfrag[tn], (short)0, c[tm][tn], false, false);
            }
        }

        if (ks + 1 < KSTEPS) {
            WAIT_ASYNC();                    // next panel landed
            __syncthreads();                 // + this wave's ds reads done
        }
    }
#undef STAGE

    // Fused epilogue: l2 -> exp(-l2/2) -> e + e^2 (covers both kernel widths).
    float acc = 0.0f;
    if (!mixed) {
        #pragma unroll
        for (int tm = 0; tm < 4; ++tm) {
            #pragma unroll
            for (int tn = 0; tn < 4; ++tn) {
                const int jl = waveN * 64 + tn * 16 + lr;
                const float nb = nBs[jl];
                #pragma unroll
                for (int v = 0; v < 8; ++v) {
                    const int il = waveM * 64 + tm * 16 + v + 8 * hi;  // C: M=v+8*hi
                    const float l2 = nAs[il] + nb - 2.0f * c[tm][tn][v];
                    const float e = __expf(-0.5f * l2);
                    acc += e + e * e;
                }
            }
        }
    } else {                                 // diagonal-straddling block
        #pragma unroll
        for (int tm = 0; tm < 4; ++tm) {
            #pragma unroll
            for (int tn = 0; tn < 4; ++tn) {
                const int jl = waveN * 64 + tn * 16 + lr;
                const float nb = nBs[jl];
                #pragma unroll
                for (int v = 0; v < 8; ++v) {
                    const int il = waveM * 64 + tm * 16 + v + 8 * hi;
                    const float l2 = nAs[il] + nb - 2.0f * c[tm][tn][v];
                    const float e = __expf(-0.5f * l2);
                    const int gi = br * BM + il;
                    const int gj = bc * BN + jl;
                    const float welem = (gi < gj) ? 2.0f : ((gi == gj) ? 1.0f : 0.0f);
                    acc += welem * (e + e * e);
                }
            }
        }
    }
    #pragma unroll
    for (int m = 16; m > 0; m >>= 1) acc += __shfl_xor(acc, m, 32);
    if (lane == 0) red[wid] = acc;
    __syncthreads();
    if (tid == 0) {
        float s = 0.0f;
        #pragma unroll
        for (int i = 0; i < 8; ++i) s += red[i];
        partials[pidx] = wblk * s;
    }
}

// -------------------------------------------------------------------------
// Kernel 3: deterministic final reduction (double accumulation).
// -------------------------------------------------------------------------
__global__ __launch_bounds__(256)
void mmd_finalize_kernel(const float* __restrict__ partials, float* __restrict__ out) {
    __shared__ double red[256];
    double s = 0.0;
    for (int i = threadIdx.x; i < NPARTIAL; i += 256) s += (double)partials[i];
    red[threadIdx.x] = s;
    __syncthreads();
    #pragma unroll
    for (int stride = 128; stride > 0; stride >>= 1) {
        if (threadIdx.x < stride) red[threadIdx.x] += red[threadIdx.x + stride];
        __syncthreads();
    }
    if (threadIdx.x == 0)
        out[0] = (float)(red[0] / ((double)N_ROWS * (double)N_ROWS));
}

extern "C" void kernel_launch(void* const* d_in, const int* in_sizes, int n_in,
                              void* d_out, int out_size, void* d_ws, size_t ws_size,
                              hipStream_t stream) {
    (void)in_sizes; (void)n_in; (void)out_size; (void)ws_size;
    const float* src = (const float*)d_in[0];
    const float* tgt = (const float*)d_in[1];

    // workspace layout
    char* ws = (char*)d_ws;
    _Float16* Sh = (_Float16*)ws;                                  // 4 MB
    _Float16* Th = Sh + (size_t)N_ROWS * DIM;                      // 4 MB
    float* ns = (float*)(Th + (size_t)N_ROWS * DIM);               // 32 KB
    float* nt = ns + N_ROWS;                                       // 32 KB
    float* partials = nt + N_ROWS;                                 // 24 KB

    mmd_prep_kernel<<<dim3(N_ROWS / 8, 2), 256, 0, stream>>>(src, tgt, Sh, Th, ns, nt);
    mmd_main_kernel<<<dim3(TM, TN, 3), 256, 0, stream>>>(Sh, Th, ns, nt, partials);
    mmd_finalize_kernel<<<1, 256, 0, stream>>>(partials, (float*)d_out);
}
